// FrameAugment_48507360641703
// MI455X (gfx1250) — matrix operand order repaired and checked by reading, not compiled, not run on hardware
//
#include <hip/hip_runtime.h>
#include <hip/hip_bf16.h>
#include <math.h>

// Problem sizes (fixed by the reference)
#define BDIM 8
#define SDIM 2048
#define FDIM 512

typedef __attribute__((ext_vector_type(16))) _Float16 v16h;
typedef __attribute__((ext_vector_type(8)))  _Float16 v8h;
typedef __attribute__((ext_vector_type(8)))  float    v8f;

union V16H { v16h v; v8h h[2]; };

// -----------------------------------------------------------------------------
// Kernel 1: row softmax of (mu + softplus(sigma_param) * noise), output f16.
// One 256-thread block per row (2048 elements -> 8 per thread).
// -----------------------------------------------------------------------------
__global__ __launch_bounds__(256) void softmax_rows_f16(
    const float* __restrict__ noise, const float* __restrict__ mu,
    const float* __restrict__ sigma_param, _Float16* __restrict__ P)
{
    __shared__ float red[256];
    const int tid = threadIdx.x;
    const size_t row = blockIdx.x;
    const float* __restrict__ x = noise + row * (size_t)SDIM;

    const float m0    = mu[0];
    const float sp    = sigma_param[0];
    const float sigma = log1pf(__expf(sp));   // softplus

    float v[SDIM / 256];
    float vmax = -3.402823466e38f;
#pragma unroll
    for (int i = 0; i < SDIM / 256; ++i) {
        float t = fmaf(sigma, x[tid + i * 256], m0);
        v[i] = t;
        vmax = fmaxf(vmax, t);
    }
    red[tid] = vmax;
    __syncthreads();
    for (int s = 128; s > 0; s >>= 1) {
        if (tid < s) red[tid] = fmaxf(red[tid], red[tid + s]);
        __syncthreads();
    }
    vmax = red[0];
    __syncthreads();

    float sum = 0.f;
#pragma unroll
    for (int i = 0; i < SDIM / 256; ++i) {
        float e = __expf(v[i] - vmax);
        v[i] = e;
        sum += e;
    }
    red[tid] = sum;
    __syncthreads();
    for (int s = 128; s > 0; s >>= 1) {
        if (tid < s) red[tid] += red[tid + s];
        __syncthreads();
    }
    const float inv = 1.0f / red[0];

    _Float16* __restrict__ o = P + row * (size_t)SDIM;
#pragma unroll
    for (int i = 0; i < SDIM / 256; ++i)
        o[tid + i * 256] = (_Float16)(v[i] * inv);
}

// -----------------------------------------------------------------------------
// Kernel 2: repack feature f32 -> f16 in WMMA B-operand tile order.
// Tiles are 32(K) x 16(N). Within a tile (512 halves):
//   lane = n%16 + 16*(k%32 >= 16), per-lane halves ordered by ascending K
//   (lanes 0-15 hold K=0..15, lanes 16-31 hold K=16..31 of the tile).
// Tile order: nTile-major, kTile contiguous:
//   tileIdx = nTile*(SDIM/32) + kTile
// -----------------------------------------------------------------------------
__global__ __launch_bounds__(256) void pack_feature_f16(
    const float* __restrict__ feat, _Float16* __restrict__ Fp)
{
    const size_t total = (size_t)BDIM * SDIM * FDIM;
    size_t idx = (size_t)blockIdx.x * blockDim.x + threadIdx.x;
    if (idx >= total) return;

    const int n = (int)(idx % FDIM);
    const size_t t = idx / FDIM;
    const int k = (int)(t % SDIM);
    const int b = (int)(t / SDIM);

    const int kt = k >> 5;         // K tile (32-wide)
    const int kk = k & 31;
    const int nt = n >> 4;         // N tile (16-wide)
    const int nn = n & 15;

    const int lane = nn + ((kk >> 4) << 4);   // 0..31
    const int pos  = kk & 15;                 // half index within lane's 16

    const size_t out = (size_t)b * SDIM * FDIM
                     + ((size_t)nt * (SDIM / 32) + kt) * 512
                     + (size_t)lane * 16 + pos;
    Fp[out] = (_Float16)feat[idx];
}

// -----------------------------------------------------------------------------
// Kernel 3: per-batch GEMM  out[b] = P[b] (2048x2048 f16) x F[b] (2048x512 f16)
// f32 accumulate via v_wmma_f32_16x16x32_f16.
// Block: 256 threads = 8 waves; waves tiled 4(M) x 2(N).
// Wave tile: 32(M) x 64(N) = 8 accumulators (2 M-subtiles x 4 N-tiles).
// Block tile: 128(M) x 128(N).
// K-loop uses a depth-2 software pipeline (3 fragment stages in flight):
// while computing step kt, the loads for kt+1 and kt+2 are outstanding,
// so waits become deep partial s_wait_loadcnt instead of full drains.
// -----------------------------------------------------------------------------
__global__ __launch_bounds__(256) void pgemm_wmma_f16(
    const _Float16* __restrict__ P, const _Float16* __restrict__ Fp,
    float* __restrict__ out)
{
    const int lane  = threadIdx.x & 31;
    const int wave  = threadIdx.x >> 5;
    const int waveM = wave & 3;     // 0..3
    const int waveN = wave >> 2;    // 0..1
    const int b     = blockIdx.z;

    const int mBase  = blockIdx.y * 128 + waveM * 32;
    const int ntBase = blockIdx.x * 8 + waveN * 4;   // 16-wide N tiles

    // A-operand addressing (contiguous K per lane half, same order as packed B):
    // lane 0-15: M=lane,   K = {0..15}
    // lane16-31: M=lane-16,K = {16..31}
    const int kA = (lane >> 4) << 4;   // 0 or 16
    const _Float16* __restrict__ aRow0 =
        P + (size_t)b * SDIM * SDIM + (size_t)(mBase + (lane & 15)) * SDIM + kA;
    const _Float16* __restrict__ aRow1 = aRow0 + (size_t)16 * SDIM;

    // B-operand: pre-packed, lane-contiguous 16 halves per tile.
    const _Float16* __restrict__ bBase =
        Fp + (size_t)b * SDIM * FDIM + (size_t)lane * 16;
    const size_t ntStride = (size_t)(SDIM / 32) * 512;   // halves per N-tile column

#define LOAD_K(kt, A0, A1, B0, B1, B2, B3)                                   \
    do {                                                                     \
        const _Float16* ap0 = aRow0 + (kt) * 32;                             \
        const _Float16* ap1 = aRow1 + (kt) * 32;                             \
        (A0).h[0] = *(const v8h*)(ap0);                                      \
        (A0).h[1] = *(const v8h*)(ap0 + 8);                                  \
        (A1).h[0] = *(const v8h*)(ap1);                                      \
        (A1).h[1] = *(const v8h*)(ap1 + 8);                                  \
        const _Float16* bt = bBase + (size_t)(kt) * 512;                     \
        (B0) = *(const v16h*)(bt + (size_t)(ntBase + 0) * ntStride);         \
        (B1) = *(const v16h*)(bt + (size_t)(ntBase + 1) * ntStride);         \
        (B2) = *(const v16h*)(bt + (size_t)(ntBase + 2) * ntStride);         \
        (B3) = *(const v16h*)(bt + (size_t)(ntBase + 3) * ntStride);         \
    } while (0)

#define WMMA1(ACC, A, B)                                                     \
    (ACC) = __builtin_amdgcn_wmma_f32_16x16x32_f16(false, (A).v, false, (B), \
                                                   (short)0, (ACC), false, false)

#define WMMA8(A0, A1, B0, B1, B2, B3)                                        \
    do {                                                                     \
        WMMA1(acc00, A0, B0);                                                \
        WMMA1(acc10, A1, B0);                                                \
        WMMA1(acc01, A0, B1);                                                \
        WMMA1(acc11, A1, B1);                                                \
        WMMA1(acc02, A0, B2);                                                \
        WMMA1(acc12, A1, B2);                                                \
        WMMA1(acc03, A0, B3);                                                \
        WMMA1(acc13, A1, B3);                                                \
    } while (0)

    v8f acc00 = {}, acc01 = {}, acc02 = {}, acc03 = {};
    v8f acc10 = {}, acc11 = {}, acc12 = {}, acc13 = {};

    // Pipeline stages: s0 = compute, s1 = +1 in flight, s2 = +2 in flight.
    V16H a0s0, a1s0; v16h b0s0, b1s0, b2s0, b3s0;
    V16H a0s1, a1s1; v16h b0s1, b1s1, b2s1, b3s1;

    LOAD_K(0, a0s0, a1s0, b0s0, b1s0, b2s0, b3s0);
    LOAD_K(1, a0s1, a1s1, b0s1, b1s1, b2s1, b3s1);

#pragma unroll 3
    for (int kt = 0; kt < SDIM / 32 - 2; ++kt) {
        // Prefetch the A stream further ahead into near caches.
        __builtin_prefetch(aRow0 + kt * 32 + 16 * 32, 0, 3);

        V16H a0s2, a1s2; v16h b0s2, b1s2, b2s2, b3s2;
        LOAD_K(kt + 2, a0s2, a1s2, b0s2, b1s2, b2s2, b3s2);

        WMMA8(a0s0, a1s0, b0s0, b1s0, b2s0, b3s0);

        // Rotate stages (pure register renaming after unroll-3).
        a0s0 = a0s1; a1s0 = a1s1;
        b0s0 = b0s1; b1s0 = b1s1; b2s0 = b2s1; b3s0 = b3s1;
        a0s1 = a0s2; a1s1 = a1s2;
        b0s1 = b0s2; b1s1 = b1s2; b2s1 = b2s2; b3s1 = b3s2;
    }

    // Epilogue: last two K-steps.
    WMMA8(a0s0, a1s0, b0s0, b1s0, b2s0, b3s0);
    WMMA8(a0s1, a1s1, b0s1, b1s1, b2s1, b3s1);

#undef LOAD_K
#undef WMMA1
#undef WMMA8

    // C/D layout: VGPR i -> row = i + 8*(lane/16), col = lane%16.
    const int rowS = mBase + ((lane >> 4) << 3);
    const int col  = lane & 15;
    float* __restrict__ outB = out + (size_t)b * SDIM * FDIM;
#pragma unroll
    for (int i = 0; i < 8; ++i) {
        float* o0 = outB + (size_t)(rowS + i) * FDIM + col;
        o0[(ntBase + 0) * 16] = acc00[i];
        o0[(ntBase + 1) * 16] = acc01[i];
        o0[(ntBase + 2) * 16] = acc02[i];
        o0[(ntBase + 3) * 16] = acc03[i];
        float* o1 = o0 + (size_t)16 * FDIM;
        o1[(ntBase + 0) * 16] = acc10[i];
        o1[(ntBase + 1) * 16] = acc11[i];
        o1[(ntBase + 2) * 16] = acc12[i];
        o1[(ntBase + 3) * 16] = acc13[i];
    }
}

// -----------------------------------------------------------------------------
// Launch. Workspace usage (needs ~80 MiB):
//   [0, 64 MiB)      : P, f16 softmax probs, B*S*S
//   [64 MiB, 80 MiB) : Fp, f16 repacked feature, B*S*F
// -----------------------------------------------------------------------------
extern "C" void kernel_launch(void* const* d_in, const int* in_sizes, int n_in,
                              void* d_out, int out_size, void* d_ws, size_t ws_size,
                              hipStream_t stream) {
    const float* feature     = (const float*)d_in[0];
    const float* noise       = (const float*)d_in[1];
    const float* mu          = (const float*)d_in[2];
    const float* sigma_param = (const float*)d_in[3];
    float* out = (float*)d_out;

    _Float16* P  = (_Float16*)d_ws;
    _Float16* Fp = (_Float16*)((char*)d_ws +
                   (size_t)BDIM * SDIM * SDIM * sizeof(_Float16));

    // 1) softmax rows -> f16 P
    softmax_rows_f16<<<BDIM * SDIM, 256, 0, stream>>>(noise, mu, sigma_param, P);

    // 2) repack feature -> f16 WMMA-B layout
    const size_t total = (size_t)BDIM * SDIM * FDIM;
    pack_feature_f16<<<(unsigned)((total + 255) / 256), 256, 0, stream>>>(feature, Fp);

    // 3) batched WMMA GEMM
    dim3 grid(FDIM / 128, SDIM / 128, BDIM);
    pgemm_wmma_f16<<<grid, 256, 0, stream>>>(P, Fp, out);
}
